// TwistorTitansRecurrent_56959856279927
// MI455X (gfx1250) — matrix operand order repaired and checked by reading, not compiled, not run on hardware
//
#include <hip/hip_runtime.h>
#include <hip/hip_bf16.h>

// ---------------- problem constants ----------------
static constexpr int BB  = 32;    // batch
static constexpr int SS  = 512;   // seq len
static constexpr int DD  = 1024;  // input dim
static constexpr int HH  = 1024;  // hidden
static constexpr int HHh = 512;   // half hidden
static constexpr int MA  = BB * SS; // 16384 rows for batched GEMMs

// ---------------- WMMA types ----------------
typedef __attribute__((ext_vector_type(16))) __bf16 v16bf;
typedef __attribute__((ext_vector_type(8)))  __bf16 v8bf;
typedef __attribute__((ext_vector_type(8)))  float  v8f;

__device__ __forceinline__ v8f wmma_bf16(v16bf a, v16bf b, v8f c) {
    // D = A(16x32 bf16) * B(32x16 bf16) + C(16x16 f32)
    return __builtin_amdgcn_wmma_f32_16x16x32_bf16(
        false, a, false, b, (short)0, c, false, false);
}

__device__ __forceinline__ unsigned short f2bf(float f) {
    unsigned u = __builtin_bit_cast(unsigned, f);
    unsigned r = u + 0x7FFFu + ((u >> 16) & 1u);   // round-to-nearest-even
    return (unsigned short)(r >> 16);
}
__device__ __forceinline__ float bf2f(unsigned short u) {
    return __builtin_bit_cast(float, ((unsigned)u) << 16);
}
__device__ __forceinline__ float sigmoidf(float x) { return 1.0f / (1.0f + __expf(-x)); }

// Build A fragment: row-major bf16 A, lane holds row (mtile*16 + lane&15),
// elements 0..7 = K hk..hk+7, 8..15 = K hk+16..hk+23, hk = (lane>>4)*8.
__device__ __forceinline__ v16bf load_afrag(const unsigned short* __restrict__ A,
                                            int row, int K, int kbase, int lane) {
    const unsigned short* ap = A + (size_t)row * K + kbase + ((lane >> 4) * 8);
    v8bf lo = *(const v8bf*)ap;
    v8bf hi = *(const v8bf*)(ap + 16);
    return __builtin_shufflevector(lo, hi, 0,1,2,3,4,5,6,7,8,9,10,11,12,13,14,15);
}

// ---------------- weight packing ----------------
// Packed B layout: tile (tn,tk) -> base ((tn*ktiles + tk)*512); within a tile
// lane l owns 16 contiguous bf16: column = tn*16 + (l&15), K = tk*32 + (l>>4)*8 + slot (slot<8), tk*32 + 16 + (l>>4)*8 + slot-8 (slot>=8).
__global__ __launch_bounds__(256)
void pack_w_kernel(const float* __restrict__ W, unsigned short* __restrict__ out,
                   int K, int N) {
    int p = blockIdx.x * 256 + threadIdx.x;
    if (p >= K * N) return;
    int tile   = p >> 9;
    int within = p & 511;
    int lane = within >> 4;
    int slot = within & 15;
    int ktiles = K >> 5;
    int tn = tile / ktiles;
    int tk = tile % ktiles;
    int row = tk * 32 + (lane >> 4) * 8 + (slot < 8 ? slot : 16 + slot - 8);
    int col = tn * 16 + (lane & 15);
    out[p] = f2bf(W[(size_t)row * N + col]);
}

__global__ __launch_bounds__(256)
void convert_bf16_kernel(const float* __restrict__ in, unsigned short* __restrict__ out,
                         long long n) {
    long long i = (long long)blockIdx.x * 256 + threadIdx.x;
    if (i < n) out[i] = f2bf(in[i]);
}

// ---------------- big batched GEMM: C = act(A*Bp + bias) ----------------
// Block = 8 waves covering 8 M-tiles (128 rows) x one 64-column group.
// The four packed B tiles per k-step (4KB) are staged in LDS with a 2-deep
// double buffer shared by all 8 waves. Steady-state loop (no conditionals):
// issue tk+1 global loads, read all four B fragments from LDS, run 4 WMMAs,
// then store tk+1 into the other buffer. Last k-step is peeled.
// Requires: M % 128 == 0, N % 64 == 0, K % 32 == 0 (true for all launches).
__global__ __launch_bounds__(256)
void gemm_bf16_wmma(const unsigned short* __restrict__ A,
                    const unsigned short* __restrict__ Bp,
                    const float* __restrict__ bias,
                    float* __restrict__ C,
                    int M, int N, int K, int act) {
    __shared__ __align__(32) unsigned short ldsB[2][4 * 512];  // 8 KB
    (void)M;

    const int tid  = threadIdx.x;
    const int wave = tid >> 5;
    const int lane = tid & 31;
    const int nblk = N >> 6;
    const int bm   = blockIdx.x / nblk;      // 128-row block index
    const int ng   = blockIdx.x % nblk;      // 64-col group
    const int mt   = bm * 8 + wave;
    const int row  = mt * 16 + (lane & 15);
    const int ktiles = K >> 5;

    // cooperative B staging: thread t loads 16B of tile j = t/64, elem (t%64)*8
    const int jt = tid >> 6;
    const int et = (tid & 63) * 8;
    const size_t btile = (size_t)(ng * 4 + jt) * ktiles;

    // prologue: stage k-step 0
    v8bf bstage = *(const v8bf*)(Bp + btile * 512 + et);
    *(v8bf*)&ldsB[0][jt * 512 + et] = bstage;
    v16bf afrag = load_afrag(A, row, K, 0, lane);

    v8f acc[4] = {};
    int buf = 0;
    for (int tk = 0; tk < ktiles - 1; ++tk) {
        __syncthreads();                                  // buf ready; other buf free
        // in-flight loads for tk+1 (drained behind the WMMAs below)
        bstage = *(const v8bf*)(Bp + (btile + tk + 1) * 512 + et);
        v16bf afragn = load_afrag(A, row, K, (tk + 1) * 32, lane);
        // read all four B fragments first so ds waits overlap the WMMAs
        v16bf bf0 = *(const v16bf*)&ldsB[buf][0 * 512 + lane * 16];
        v16bf bf1 = *(const v16bf*)&ldsB[buf][1 * 512 + lane * 16];
        v16bf bf2 = *(const v16bf*)&ldsB[buf][2 * 512 + lane * 16];
        v16bf bf3 = *(const v16bf*)&ldsB[buf][3 * 512 + lane * 16];
        acc[0] = wmma_bf16(afrag, bf0, acc[0]);
        acc[1] = wmma_bf16(afrag, bf1, acc[1]);
        acc[2] = wmma_bf16(afrag, bf2, acc[2]);
        acc[3] = wmma_bf16(afrag, bf3, acc[3]);
        buf ^= 1;
        *(v8bf*)&ldsB[buf][jt * 512 + et] = bstage;       // waits global here
        afrag = afragn;
    }
    __syncthreads();
    {
        v16bf bf0 = *(const v16bf*)&ldsB[buf][0 * 512 + lane * 16];
        v16bf bf1 = *(const v16bf*)&ldsB[buf][1 * 512 + lane * 16];
        v16bf bf2 = *(const v16bf*)&ldsB[buf][2 * 512 + lane * 16];
        v16bf bf3 = *(const v16bf*)&ldsB[buf][3 * 512 + lane * 16];
        acc[0] = wmma_bf16(afrag, bf0, acc[0]);
        acc[1] = wmma_bf16(afrag, bf1, acc[1]);
        acc[2] = wmma_bf16(afrag, bf2, acc[2]);
        acc[3] = wmma_bf16(afrag, bf3, acc[3]);
    }
#pragma unroll
    for (int j = 0; j < 4; ++j) {
#pragma unroll
        for (int g = 0; g < 8; ++g) {
            int r = mt * 16 + g + (lane >> 4) * 8;   // C layout: VGPR g = rows g / g+8
            int c = (ng * 4 + j) * 16 + (lane & 15);
            float v = acc[j][g] + bias[c];
            if (act == 1) v = tanhf(v);
            C[(size_t)r * N + c] = v;
        }
    }
}

// ---------------- per-step GEMM: cand-pre (W_h) + r-pre (W_r_h) + svals ----
// blocks 0..7: 64 waves = {2 matrices} x {2 mtiles} x {16 ngroups of 64 cols}
// block 8    : 96 scalar dots for s = h@W_s + b_s
// Register double buffering with a peeled last iteration: fragments for tk+1
// are in flight while the four WMMAs of tk execute; no per-iteration branches.
__global__ __launch_bounds__(256)
void step_gemm_kernel(const unsigned short* __restrict__ hcat,   // 32x1024 bf16
                      const unsigned short* __restrict__ WhP,
                      const unsigned short* __restrict__ WrhP,
                      const float* __restrict__ xproj,           // (B*S)xH (tanh'd, biased)
                      const float* __restrict__ xr,              // (B*S)xH (biased w/ b_r)
                      const float* __restrict__ b_h,
                      const float* __restrict__ W_s,             // Hx3
                      const float* __restrict__ b_s,
                      float* __restrict__ cand,                  // 32x1024
                      float* __restrict__ rbuf,                  // 32x1024
                      float* __restrict__ svals,                 // 32x3
                      int t) {
    if (blockIdx.x == 8) {
        int id = threadIdx.x;
        if (id < BB * 3) {
            int b = id / 3, c = id % 3;
            float s = b_s[c];
            for (int k = 0; k < HH; ++k)
                s = fmaf(bf2f(hcat[b * HH + k]), W_s[k * 3 + c], s);
            svals[b * 3 + c] = s;
        }
        return;
    }
    const int wave = (blockIdx.x * 256 + threadIdx.x) >> 5;  // 0..63
    const int lane = threadIdx.x & 31;
    const int mat  = wave >> 5;            // 0 = W_h, 1 = W_r_h
    const int wv   = wave & 31;
    const int mt   = wv >> 4;              // 0..1
    const int ng   = wv & 15;              // 0..15
    const unsigned short* Bp = (mat == 0) ? WhP : WrhP;
    const int row = mt * 16 + (lane & 15);
    const int ktiles = HH >> 5;            // 32

    auto bload = [&](int tk, int j) -> v16bf {
        const unsigned short* bp = Bp + ((size_t)(ng * 4 + j) * ktiles + tk) * 512 + lane * 16;
        return *(const v16bf*)bp;
    };

    v16bf afrag = load_afrag(hcat, row, HH, 0, lane);
    v16bf bfr[4];
#pragma unroll
    for (int j = 0; j < 4; ++j) bfr[j] = bload(0, j);

    v8f acc[4] = {};
    for (int tk = 0; tk < ktiles - 1; ++tk) {
        // in-flight during the WMMAs below
        v16bf afragn = load_afrag(hcat, row, HH, (tk + 1) * 32, lane);
        v16bf bfrn[4];
#pragma unroll
        for (int j = 0; j < 4; ++j) bfrn[j] = bload(tk + 1, j);
#pragma unroll
        for (int j = 0; j < 4; ++j)
            acc[j] = wmma_bf16(afrag, bfr[j], acc[j]);
        afrag = afragn;
#pragma unroll
        for (int j = 0; j < 4; ++j) bfr[j] = bfrn[j];
    }
#pragma unroll
    for (int j = 0; j < 4; ++j)
        acc[j] = wmma_bf16(afrag, bfr[j], acc[j]);

#pragma unroll
    for (int j = 0; j < 4; ++j) {
#pragma unroll
        for (int g = 0; g < 8; ++g) {
            int b = mt * 16 + g + (lane >> 4) * 8;        // batch row, 0..31
            int c = (ng * 4 + j) * 16 + (lane & 15);      // 0..1023
            size_t xoff = ((size_t)b * SS + t) * HH + c;
            if (mat == 0) {
                cand[b * HH + c] = tanhf(acc[j][g] + xproj[xoff] + b_h[c]);
            } else {
                rbuf[b * HH + c] = sigmoidf(acc[j][g] + xr[xoff]);  // b_r folded into xr
            }
        }
    }
}

// ---------------- per-step elementwise update ----------------
__global__ __launch_bounds__(256)
void step_update_kernel(const float* __restrict__ cand, const float* __restrict__ rbuf,
                        const float* __restrict__ svals,
                        const float* __restrict__ mob_a, const float* __restrict__ mob_b,
                        const float* __restrict__ mob_c, const float* __restrict__ mob_d,
                        float* __restrict__ homega, float* __restrict__ hpi,
                        unsigned short* __restrict__ hcat,
                        unsigned short* __restrict__ hstates, int t) {
    int idx = blockIdx.x * 256 + threadIdx.x;     // 64 blocks -> 16384 = B*Hh
    int b = idx >> 9;
    int i = idx & (HHh - 1);
    float zo = cand[b * HH + i];
    float zp = cand[b * HH + HHh + i];
#pragma unroll
    for (int k = 0; k < 3; ++k) {
        float a_re = mob_a[(k * 2 + 0) * HHh + i], a_im = mob_a[(k * 2 + 1) * HHh + i];
        float b_re = mob_b[(k * 2 + 0) * HHh + i], b_im = mob_b[(k * 2 + 1) * HHh + i];
        float c_re = mob_c[(k * 2 + 0) * HHh + i], c_im = mob_c[(k * 2 + 1) * HHh + i];
        float d_re = mob_d[(k * 2 + 0) * HHh + i], d_im = mob_d[(k * 2 + 1) * HHh + i];
        float num_re = a_re * zo - a_im * zp + b_re;
        float num_im = a_re * zp + a_im * zo + b_im;
        float den_re = c_re * zo - c_im * zp + d_re;
        float den_im = c_re * zp + c_im * zo + d_im;
        float inv = 1.0f / (den_re * den_re + den_im * den_im + 1e-6f);
        float nzo = (num_re * den_re + num_im * den_im) * inv;
        float nzp = (num_im * den_re - num_re * den_im) * inv;
        zo = nzo; zp = nzp;
    }
    float s0 = svals[b * 3 + 0], s1 = svals[b * 3 + 1], s2 = svals[b * 3 + 2];
    float m = fmaxf(s0, fmaxf(s1, s2));
    float e0 = __expf(s0 - m), e1 = __expf(s1 - m), e2 = __expf(s2 - m);
    float wsc = (e0 + 0.5f * e1 + 0.25f * e2) / (e0 + e1 + e2);
    float ro = rbuf[b * HH + i] * wsc;
    float rp = rbuf[b * HH + HHh + i] * wsc;
    float no = (1.0f - ro) * homega[b * HHh + i] + ro * zo;
    float np = (1.0f - rp) * hpi[b * HHh + i]    + rp * zp;
    homega[b * HHh + i] = no;
    hpi[b * HHh + i]    = np;
    unsigned short nob = f2bf(no), npb = f2bf(np);
    hcat[b * HH + i]        = nob;
    hcat[b * HH + HHh + i]  = npb;
    size_t base = ((size_t)b * SS + t) * HH;
    hstates[base + i]        = nob;
    hstates[base + HHh + i]  = npb;
}

__global__ __launch_bounds__(256)
void init_state_kernel(float* __restrict__ homega, float* __restrict__ hpi,
                       unsigned short* __restrict__ hcat) {
    int idx = blockIdx.x * 256 + threadIdx.x;
    if (idx < BB * HH) hcat[idx] = 0;
    if (idx < BB * HHh) { homega[idx] = 0.0f; hpi[idx] = 0.0f; }
}

__global__ __launch_bounds__(256)
void copy_final_kernel(const float* __restrict__ homega, const float* __restrict__ hpi,
                       float* __restrict__ dst) {
    int idx = blockIdx.x * 256 + threadIdx.x;
    if (idx < BB * HHh) {
        dst[idx] = homega[idx];
        dst[BB * HHh + idx] = hpi[idx];
    }
}

// ---------------- host orchestration ----------------
extern "C" void kernel_launch(void* const* d_in, const int* in_sizes, int n_in,
                              void* d_out, int out_size, void* d_ws, size_t ws_size,
                              hipStream_t stream) {
    (void)in_sizes; (void)n_in; (void)out_size; (void)ws_size;
    const float* x     = (const float*)d_in[0];
    const float* W_x   = (const float*)d_in[1];
    const float* b_x   = (const float*)d_in[2];
    const float* W_h   = (const float*)d_in[3];
    const float* b_h   = (const float*)d_in[4];
    const float* W_r   = (const float*)d_in[5];   // (D+H, H)
    const float* b_r   = (const float*)d_in[6];
    const float* W_s   = (const float*)d_in[7];
    const float* b_s   = (const float*)d_in[8];
    const float* mob_a = (const float*)d_in[9];
    const float* mob_b = (const float*)d_in[10];
    const float* mob_c = (const float*)d_in[11];
    const float* mob_d = (const float*)d_in[12];
    const float* W_o   = (const float*)d_in[13];
    const float* b_o   = (const float*)d_in[14];
    float* out = (float*)d_out;

    // workspace carve-up (256B aligned)
    char* ws = (char*)d_ws;
    size_t o = 0;
    auto carve = [&](size_t bytes) { char* p = ws + o; o += (bytes + 255) & ~(size_t)255; return p; };
    unsigned short* xb      = (unsigned short*)carve((size_t)MA * DD * 2);
    unsigned short* hstates = (unsigned short*)carve((size_t)MA * HH * 2);
    unsigned short* wx_p    = (unsigned short*)carve((size_t)DD * HH * 2);
    unsigned short* wh_p    = (unsigned short*)carve((size_t)HH * HH * 2);
    unsigned short* wrx_p   = (unsigned short*)carve((size_t)DD * HH * 2);
    unsigned short* wrh_p   = (unsigned short*)carve((size_t)HH * HH * 2);
    unsigned short* wo_p    = (unsigned short*)carve((size_t)HH * HH * 2);
    float*          xproj   = (float*)carve((size_t)MA * HH * 4);
    float*          xr      = (float*)carve((size_t)MA * HH * 4);
    float*          cand    = (float*)carve((size_t)BB * HH * 4);
    float*          rbuf    = (float*)carve((size_t)BB * HH * 4);
    float*          svals   = (float*)carve(512);
    float*          homega  = (float*)carve((size_t)BB * HHh * 4);
    float*          hpi     = (float*)carve((size_t)BB * HHh * 4);
    unsigned short* hcat    = (unsigned short*)carve((size_t)BB * HH * 2);

    const int KN = HH * HH;
    const int pkB = (KN + 255) / 256;
    // pack weights into WMMA-B tile layout (bf16)
    pack_w_kernel<<<pkB, 256, 0, stream>>>(W_x, wx_p, DD, HH);
    pack_w_kernel<<<pkB, 256, 0, stream>>>(W_h, wh_p, HH, HH);
    pack_w_kernel<<<pkB, 256, 0, stream>>>(W_r, wrx_p, DD, HH);                   // rows [0,D)
    pack_w_kernel<<<pkB, 256, 0, stream>>>(W_r + (size_t)DD * HH, wrh_p, HH, HH); // rows [D,D+H)
    pack_w_kernel<<<pkB, 256, 0, stream>>>(W_o, wo_p, HH, HH);

    long long nx = (long long)MA * DD;
    convert_bf16_kernel<<<(int)((nx + 255) / 256), 256, 0, stream>>>(x, xb, nx);

    // batched input projections (off the critical path)
    const int blocksA = (MA / 128) * (HH / 64);   // 2048 blocks, 8 waves each
    gemm_bf16_wmma<<<blocksA, 256, 0, stream>>>(xb, wx_p,  b_x, xproj, MA, HH, DD, 1); // tanh
    gemm_bf16_wmma<<<blocksA, 256, 0, stream>>>(xb, wrx_p, b_r, xr,    MA, HH, DD, 0); // +b_r, linear

    init_state_kernel<<<(BB * HH + 255) / 256, 256, 0, stream>>>(homega, hpi, hcat);

    // sequential recurrence: two skinny WMMA GEMMs + elementwise update per step
    for (int t = 0; t < SS; ++t) {
        step_gemm_kernel<<<9, 256, 0, stream>>>(hcat, wh_p, wrh_p, xproj, xr,
                                                b_h, W_s, b_s, cand, rbuf, svals, t);
        step_update_kernel<<<(BB * HHh) / 256, 256, 0, stream>>>(
            cand, rbuf, svals, mob_a, mob_b, mob_c, mob_d,
            homega, hpi, hcat, hstates, t);
    }

    // batched output projection straight into d_out
    gemm_bf16_wmma<<<blocksA, 256, 0, stream>>>(hstates, wo_p, b_o, out, MA, HH, HH, 1);
    copy_final_kernel<<<(BB * HHh + 255) / 256, 256, 0, stream>>>(
        homega, hpi, out + (size_t)MA * HH);
}
